// UnitNet_18373870092889
// MI455X (gfx1250) — matrix operand order repaired and checked by reading, not compiled, not run on hardware
//
#include <hip/hip_runtime.h>
#include <hip/hip_bf16.h>

typedef __attribute__((ext_vector_type(16))) _Float16 v16h;
typedef __attribute__((ext_vector_type(8)))  _Float16 v8h;
typedef __attribute__((ext_vector_type(8)))  float    v8f;

#if defined(__gfx1250__)
#define ASYNC_LDS 1
#endif

// ---------------------------------------------------------------------------
// Implicit-GEMM conv3x3 via V_WMMA_F32_16X16X32_F16, fp16 NHWC activations.
// Each wave: 16 pixels x 32 couts (two accumulators) -> each A fragment feeds
// two WMMAs, halving global A traffic per FLOP.
// A-fragment (16x32 f16): lane half h holds K=16h+0..15 (same order as B)
//   -> two b128 loads from channel-contiguous NHWC memory.
// B-fragment (32x16 f16): lane needs K=16h+0..15 contiguous -> 16B LDS reads
//   from the 32-cout weight slice staged in LDS (ASYNCcnt-tracked DMA).
// Epilogue: +bias, optional leaky-relu, optional residual add.
//   mode 0: fp16 NHWC out/add.  mode 1: fp32 NCHW out/add (final layer).
// ---------------------------------------------------------------------------
template<int CIN, int KS>
__global__ __launch_bounds__(128)
void conv_wmma(const _Float16* __restrict__ in, int Hin, int Win,
               const _Float16* __restrict__ wgt,      // fp16 [KS*KS][Coutp][CIN]
               const float* __restrict__ bias,
               void* __restrict__ outv, int Cout, int Coutp,
               int Hout, int Wout, int stride, int act,
               const void* __restrict__ addv, int mode)
{
  constexpr int PAD = (KS - 1) / 2;
  constexpr int CH  = CIN / 8;                        // v8h chunks per row
  __shared__ _Float16 w_s[KS * KS * 32 * CIN];

  const int HW  = Hout * Wout;
  const int cob = blockIdx.y * 32;
  const bool two = (cob + 16) < Coutp;

  // stage this block's 32-cout weight slice into LDS (async global->LDS DMA)
  for (int j = threadIdx.x; j < KS * KS * 32 * CH; j += 128) {
    const int q    = j % CH;
    const int rr   = j / CH;
    const int co32 = rr & 31;
    const int t    = rr >> 5;
    _Float16* dst = &w_s[(size_t)(t * 32 + co32) * CIN + q * 8];
    if (cob + co32 < Coutp) {
      const _Float16* src = &wgt[((size_t)t * Coutp + cob + co32) * CIN + q * 8];
#if ASYNC_LDS
      // GV mode: vdst = LDS byte address (low 32 bits of shared pointer),
      // vaddr = 64-bit global address.  Tracked by ASYNCcnt (ISA 15.18.3 op 98).
      asm volatile("global_load_async_to_lds_b128 %0, %1, off"
                   :: "v"((unsigned)(uintptr_t)dst), "v"(src)
                   : "memory");
#else
      *(v8h*)dst = *(const v8h*)src;
#endif
    } else {
      v8h z;
#pragma unroll
      for (int e = 0; e < 8; ++e) z[e] = (_Float16)0.f;
      *(v8h*)dst = z;
    }
  }
#if ASYNC_LDS
  asm volatile("s_wait_asynccnt 0x0" ::: "memory");
#endif
  __syncthreads();

  const int lane = threadIdx.x & 31;
  const int wave = threadIdx.x >> 5;
  const int tile = blockIdx.x * 4 + wave;
  if (tile * 16 >= HW) return;                        // wave-uniform, EXEC all-ones

  const int half = lane >> 4;
  const int nl   = lane & 15;

  v8f acc0, acc1;
  {
    const int c0 = cob + nl, c1 = cob + 16 + nl;
    const float b0 = (c0 < Cout) ? bias[c0] : 0.f;
    const float b1 = (c1 < Cout) ? bias[c1] : 0.f;
#pragma unroll
    for (int r = 0; r < 8; ++r) { acc0[r] = b0; acc1[r] = b1; }
  }

  const int p    = tile * 16 + nl;                    // A row M = lane%16
  const bool pin = p < HW;
  const int ho = p / Wout, wo = p % Wout;
  if (pin) __builtin_prefetch(in + ((size_t)(ho * stride) * Win + wo * stride) * CIN, 0, 0);

#pragma unroll
  for (int t = 0; t < KS * KS; ++t) {
    const int ky = t / KS, kx = t % KS;
    const int hi = ho * stride + ky - PAD;
    const int wi = wo * stride + kx - PAD;
    const bool inb = pin && hi >= 0 && hi < Hin && wi >= 0 && wi < Win;
    const _Float16* ap = in + ((size_t)hi * Win + wi) * CIN;

#pragma unroll
    for (int kc = 0; kc < CIN; kc += 32) {
      v16h a;
      if (inb) {
        const v8h l0 = *(const v8h*)(ap + kc + 16 * half);
        const v8h l1 = *(const v8h*)(ap + kc + 16 * half + 8);
#pragma unroll
        for (int e = 0; e < 8; ++e) { a[e] = l0[e]; a[e + 8] = l1[e]; }
      } else {
#pragma unroll
        for (int e = 0; e < 16; ++e) a[e] = (_Float16)0.f;
      }
      {
        const _Float16* wp = &w_s[(size_t)(t * 32 + nl) * CIN + kc + 16 * half];
        const v8h w0 = *(const v8h*)wp;
        const v8h w1 = *(const v8h*)(wp + 8);
        v16h b;
#pragma unroll
        for (int e = 0; e < 8; ++e) { b[e] = w0[e]; b[e + 8] = w1[e]; }
        acc0 = __builtin_amdgcn_wmma_f32_16x16x32_f16(false, a, false, b,
                                                      (short)0, acc0, false, false);
      }
      if (two) {
        const _Float16* wp = &w_s[(size_t)(t * 32 + 16 + nl) * CIN + kc + 16 * half];
        const v8h w0 = *(const v8h*)wp;
        const v8h w1 = *(const v8h*)(wp + 8);
        v16h b;
#pragma unroll
        for (int e = 0; e < 8; ++e) { b[e] = w0[e]; b[e + 8] = w1[e]; }
        acc1 = __builtin_amdgcn_wmma_f32_16x16x32_f16(false, a, false, b,
                                                      (short)0, acc1, false, false);
      }
    }
  }

#pragma unroll
  for (int s = 0; s < 2; ++s) {
    if (s == 1 && !two) break;
    const v8f& ac = s ? acc1 : acc0;
    const int co = cob + s * 16 + nl;
    if (mode == 0) {
      _Float16* out = (_Float16*)outv;
      const _Float16* add = (const _Float16*)addv;
#pragma unroll
      for (int r = 0; r < 8; ++r) {                   // D: lane holds N, M=r+8*half
        const int M = r + 8 * half, pp = tile * 16 + M;
        if (pp < HW && co < Cout) {
          float y = ac[r];
          if (act) y = (y >= 0.f) ? y : 0.2f * y;
          if (add) y += (float)add[(size_t)pp * Cout + co];
          out[(size_t)pp * Cout + co] = (_Float16)y;
        }
      }
    } else {                                          // final layer: fp32 NCHW + x
      float* out = (float*)outv;
      const float* add = (const float*)addv;
#pragma unroll
      for (int r = 0; r < 8; ++r) {
        const int M = r + 8 * half, pp = tile * 16 + M;
        if (pp < HW && co < Cout) {
          float y = ac[r];
          if (act) y = (y >= 0.f) ? y : 0.2f * y;
          if (add) y += add[(size_t)co * HW + pp];
          out[(size_t)co * HW + pp] = y;
        }
      }
    }
  }
}

// ---------------------------------------------------------------------------
// Fused kconv_ker (1x1 conv, Cout=25C) + per-pixel 5x5 dynamic filtering.
// Phase 1: A fragments (per-pixel kf features) hoisted & reused across all
//          25C/16 column tiles; 8 waves WMMA the 16x(25C) taps into LDS.
// Phase 2: edge-padded 5x5 dynamic filter, channel-fastest for coalescing.
// ---------------------------------------------------------------------------
template<int C>
__global__ __launch_bounds__(256)
void fac_fused(const _Float16* __restrict__ feat, const _Float16* __restrict__ kf,
               const _Float16* __restrict__ wk,   // fp16 [25C][C]
               const float* __restrict__ bk,
               const _Float16* __restrict__ skip, _Float16* __restrict__ out,
               int H, int W)
{
  extern __shared__ float s_k[];                      // [16 pixels][25C taps]
  constexpr int KC = 25 * C;
  constexpr int NK = C / 32;
  const int HW   = H * W;
  const int tile = blockIdx.x;
  const int lane = threadIdx.x & 31;
  const int wave = threadIdx.x >> 5;
  const int half = lane >> 4;
  const int nl   = lane & 15;

  const int p    = tile * 16 + nl;
  const bool pin = p < HW;
  const _Float16* ap = kf + (size_t)p * C;

  v16h afr[NK];                                       // hoisted A fragments
#pragma unroll
  for (int k = 0; k < NK; ++k) {
    if (pin) {
      const v8h l0 = *(const v8h*)(ap + k * 32 + 16 * half);
      const v8h l1 = *(const v8h*)(ap + k * 32 + 16 * half + 8);
#pragma unroll
      for (int e = 0; e < 8; ++e) { afr[k][e] = l0[e]; afr[k][e + 8] = l1[e]; }
    } else {
#pragma unroll
      for (int e = 0; e < 16; ++e) afr[k][e] = (_Float16)0.f;
    }
  }

  for (int n = wave; n < KC / 16; n += 8) {
    const int co = n * 16 + nl;
    v8f acc;
    const float bv = bk[co];
#pragma unroll
    for (int r = 0; r < 8; ++r) acc[r] = bv;

#pragma unroll
    for (int k = 0; k < NK; ++k) {
      const _Float16* wp = wk + (size_t)co * C + k * 32 + 16 * half;
      const v8h w0 = *(const v8h*)wp;
      const v8h w1 = *(const v8h*)(wp + 8);
      v16h b;
#pragma unroll
      for (int e = 0; e < 8; ++e) { b[e] = w0[e]; b[e + 8] = w1[e]; }
      acc = __builtin_amdgcn_wmma_f32_16x16x32_f16(false, afr[k], false, b,
                                                   (short)0, acc, false, false);
    }
#pragma unroll
    for (int r = 0; r < 8; ++r) s_k[(size_t)(r + 8 * half) * KC + co] = acc[r];
  }
  __syncthreads();

  for (int idx = threadIdx.x; idx < 16 * C; idx += 256) {
    const int c  = idx % C;                           // channel-fastest -> coalesced
    const int m  = idx / C;
    const int pp = tile * 16 + m;
    if (pp >= HW) continue;
    const int h = pp / W, w = pp % W;
    const float* kr = s_k + (size_t)m * KC + c * 25;
    float s = 0.f;
#pragma unroll
    for (int i = 0; i < 5; ++i) {
      int hh = h + i - 2; hh = hh < 0 ? 0 : (hh >= H ? H - 1 : hh);
#pragma unroll
      for (int j = 0; j < 5; ++j) {
        int ww = w + j - 2; ww = ww < 0 ? 0 : (ww >= W ? W - 1 : ww);
        s += (float)feat[((size_t)hh * W + ww) * C + c] * kr[i * 5 + j];
      }
    }
    if (skip) s += (float)skip[(size_t)pp * C + c];
    out[(size_t)pp * C + c] = (_Float16)s;
  }
}

// x fp32 NCHW [4][HW] -> fp16 NHWC [HW][32], channels 4..31 zero-padded
__global__ void cvt_pad_input(const float* __restrict__ x, _Float16* __restrict__ out, int HW)
{
  size_t idx = (size_t)blockIdx.x * blockDim.x + threadIdx.x;
  if (idx >= (size_t)HW * 32) return;
  const int c = (int)(idx & 31);
  const size_t p = idx >> 5;
  out[idx] = (_Float16)((c < 4) ? x[(size_t)c * HW + p] : 0.f);
}

// fp32 OIHW [Cout][Cin][ks][ks] -> fp16 [ks*ks][Coutp][CinP], zero-padded
__global__ void wprep(const float* __restrict__ w, _Float16* __restrict__ out,
                      int Cout, int Coutp, int Cin, int CinP, int ks)
{
  const size_t total = (size_t)ks * ks * Coutp * CinP;
  size_t idx = (size_t)blockIdx.x * blockDim.x + threadIdx.x;
  if (idx >= total) return;
  const int ci = (int)(idx % CinP);
  size_t t2 = idx / CinP;
  const int co = (int)(t2 % Coutp);
  const int t  = (int)(t2 / Coutp);
  const int ky = t / ks, kx = t % ks;
  const float v = (co < Cout && ci < Cin)
                ? w[(((size_t)co * Cin + ci) * ks + ky) * ks + kx] : 0.f;
  out[idx] = (_Float16)v;
}

// pixel-shuffle(r=2)+lrelu, NHWC fp16: in [H][W][4C] -> out [2H][2W][C]
__global__ void pixshuf_lrelu(const _Float16* __restrict__ in, _Float16* __restrict__ out,
                              int C, int H, int W)
{
  const size_t total = (size_t)4 * C * H * W;
  size_t idx = (size_t)blockIdx.x * blockDim.x + threadIdx.x;
  if (idx >= total) return;
  const int cin4 = (int)(idx % (4 * C));
  const size_t p = idx / (4 * C);
  const int w = (int)(p % W), h = (int)(p / W);
  const int c = cin4 >> 2, r = (cin4 >> 1) & 1, s = cin4 & 1;
  float v = (float)in[idx];
  v = (v >= 0.f) ? v : 0.2f * v;
  out[(((size_t)(2 * h + r)) * (2 * W) + (2 * w + s)) * C + c] = (_Float16)v;
}

// ---------------------------------------------------------------------------
// Host orchestration
// ---------------------------------------------------------------------------
static void conv3(hipStream_t st, const _Float16* in, int CinP, int Hin, int Win,
                  const _Float16* wf, const float* bias, void* out,
                  int Cout, int Coutp, int stride, int act, const void* add, int mode)
{
  const int Hout = Hin / stride, Wout = Win / stride;
  const int tiles = (Hout * Wout + 15) / 16;
  dim3 g((tiles + 3) / 4, (Coutp + 31) / 32);
  if (CinP == 32)
    conv_wmma<32, 3><<<g, 128, 0, st>>>(in, Hin, Win, wf, bias, out, Cout, Coutp, Hout, Wout, stride, act, add, mode);
  else if (CinP == 64)
    conv_wmma<64, 3><<<g, 128, 0, st>>>(in, Hin, Win, wf, bias, out, Cout, Coutp, Hout, Wout, stride, act, add, mode);
  else
    conv_wmma<128, 3><<<g, 128, 0, st>>>(in, Hin, Win, wf, bias, out, Cout, Coutp, Hout, Wout, stride, act, add, mode);
}

extern "C" void kernel_launch(void* const* d_in, const int* in_sizes, int n_in,
                              void* d_out, int out_size, void* d_ws, size_t ws_size,
                              hipStream_t stream)
{
  (void)in_sizes; (void)n_in; (void)out_size; (void)ws_size;
  const float* x = (const float*)d_in[0];
  auto P = [&](int i) { return (const float*)d_in[i]; };

  const int H1 = 256, W1 = 256, H2 = 128, W2 = 128, H3 = 64, W3 = 64;
  const int HW1 = H1 * W1, HW2 = H2 * W2, HW3 = H3 * W3;

  _Float16* base = (_Float16*)d_ws;
  size_t off = 0;
  auto halloc = [&](size_t n) { _Float16* p = base + off; off += (n + 7) & ~(size_t)7; return p; };

  // fp16 weight prep (runs every call; tiny)
  auto wpre = [&](const float* w, int Cout, int Coutp, int Cin, int CinP, int ks) {
    const size_t n = (size_t)ks * ks * Coutp * CinP;
    _Float16* dst = halloc(n);
    wprep<<<dim3((unsigned)((n + 255) / 256)), 256, 0, stream>>>(w, dst, Cout, Coutp, Cin, CinP, ks);
    return dst;
  };
  auto resb = [&](const _Float16* in, _Float16* tmp, _Float16* outb, int iw, int C, int H, int W) {
    _Float16* w1 = wpre(P(iw), C, C, C, C, 3);
    _Float16* w2 = wpre(P(iw + 2), C, C, C, C, 3);
    conv3(stream, in, C, H, W, w1, P(iw + 1), tmp, C, C, 1, 1, nullptr, 0);
    conv3(stream, tmp, C, H, W, w2, P(iw + 3), outb, C, C, 1, 0, in, 0);
  };
  auto facL = [&](const _Float16* feat, const _Float16* kfp, int iw,
                  const _Float16* skip, _Float16* outb, int C, int H, int W) {
    _Float16* wkf = wpre(P(iw), 25 * C, 25 * C, C, C, 1);
    const size_t lds = (size_t)16 * 25 * C * sizeof(float);
    const int tiles = (H * W + 15) / 16;
    if (C == 32)       fac_fused<32><<<tiles, 256, lds, stream>>>(feat, kfp, wkf, P(iw + 1), skip, outb, H, W);
    else if (C == 64)  fac_fused<64><<<tiles, 256, lds, stream>>>(feat, kfp, wkf, P(iw + 1), skip, outb, H, W);
    else               fac_fused<128><<<tiles, 256, lds, stream>>>(feat, kfp, wkf, P(iw + 1), skip, outb, H, W);
  };
  auto pshuf = [&](const _Float16* in, _Float16* out, int C, int H, int W) {
    const size_t total = (size_t)4 * C * H * W;
    pixshuf_lrelu<<<dim3((unsigned)((total + 255) / 256)), 256, 0, stream>>>(in, out, C, H, W);
  };

  // activation arena (fp16 NHWC)
  const size_t A1 = (size_t)32 * HW1;   // also fits 128ch x HW2 pre-shuffle
  const size_t A2 = (size_t)64 * HW2;   // also fits 256ch x HW3 pre-shuffle
  const size_t A3 = (size_t)128 * HW3;
  _Float16 *s1[6], *s2[6], *s3[6];
  for (int i = 0; i < 6; ++i) s1[i] = halloc(A1);
  for (int i = 0; i < 6; ++i) s2[i] = halloc(A2);
  for (int i = 0; i < 6; ++i) s3[i] = halloc(A3);

  // input: fp32 NCHW 4ch -> fp16 NHWC 32ch (zero-padded)
  _Float16* xh = halloc((size_t)HW1 * 32);
  cvt_pad_input<<<dim3((unsigned)(((size_t)HW1 * 32 + 255) / 256)), 256, 0, stream>>>(x, xh, HW1);

  // ---- kernel-conditioning branch ----
  { _Float16* w = wpre(P(31), 32, 32, 4, 32, 3);
    conv3(stream, xh, 32, H1, W1, w, P(32), s1[0], 32, 32, 1, 1, nullptr, 0); }     // kconv_11
  resb(s1[0], s1[1], s1[2], 33, 32, H1, W1);                                        // kconv_12
  resb(s1[2], s1[1], s1[0], 37, 32, H1, W1);                                        // kconv_13
  _Float16* kf1 = s1[0];
  { _Float16* w = wpre(P(41), 64, 64, 32, 32, 3);
    conv3(stream, kf1, 32, H1, W1, w, P(42), s2[0], 64, 64, 2, 1, nullptr, 0); }    // kconv_21
  resb(s2[0], s2[1], s2[2], 43, 64, H2, W2);                                        // kconv_22
  resb(s2[2], s2[1], s2[0], 47, 64, H2, W2);                                        // kconv_23
  _Float16* kf2 = s2[0];
  { _Float16* w = wpre(P(51), 128, 128, 64, 64, 3);
    conv3(stream, kf2, 64, H2, W2, w, P(52), s3[0], 128, 128, 2, 1, nullptr, 0); }  // kconv_31
  resb(s3[0], s3[1], s3[2], 53, 128, H3, W3);                                       // kconv_32
  resb(s3[2], s3[1], s3[0], 57, 128, H3, W3);                                       // kconv_33
  _Float16* kf3 = s3[0];

  // ---- encoder ----
  { _Float16* w = wpre(P(1), 32, 32, 4, 32, 3);
    conv3(stream, xh, 32, H1, W1, w, P(2), s1[1], 32, 32, 1, 1, nullptr, 0); }      // conv_11
  resb(s1[1], s1[2], s1[3], 3, 32, H1, W1);                                         // conv_12
  resb(s1[3], s1[2], s1[1], 7, 32, H1, W1);                                         // conv_13
  _Float16* f1 = s1[1];
  { _Float16* w = wpre(P(11), 64, 64, 32, 32, 3);
    conv3(stream, f1, 32, H1, W1, w, P(12), s2[1], 64, 64, 2, 1, nullptr, 0); }     // conv_21
  resb(s2[1], s2[2], s2[3], 13, 64, H2, W2);                                        // conv_22
  resb(s2[3], s2[2], s2[1], 17, 64, H2, W2);                                        // conv_23
  _Float16* f2 = s2[1];
  { _Float16* w = wpre(P(21), 128, 128, 64, 64, 3);
    conv3(stream, f2, 64, H2, W2, w, P(22), s3[1], 128, 128, 2, 1, nullptr, 0); }   // conv_31
  resb(s3[1], s3[2], s3[3], 23, 128, H3, W3);                                       // conv_32
  resb(s3[3], s3[2], s3[1], 27, 128, H3, W3);                                       // conv_33
  _Float16* f3 = s3[1];

  // ---- decoder: scale 3 ----
  facL(f3, kf3, 65, nullptr, s3[2], 128, H3, W3);                                   // ker3 + fac
  { _Float16* w = wpre(P(67), 256, 256, 128, 128, 3);
    conv3(stream, s3[2], 128, H3, W3, w, P(68), s2[2], 256, 256, 1, 0, nullptr, 0);} // upconv_21
  pshuf(s2[2], s2[3], 64, H3, W3);                                                  // -> [HW2][64]
  resb(s2[3], s2[4], s2[5], 69, 64, H2, W2);                                        // upconv_22
  resb(s2[5], s2[4], s2[3], 73, 64, H2, W2);                                        // upconv_23

  // ---- decoder: scale 2 ----
  facL(s2[3], kf2, 63, f2, s2[2], 64, H2, W2);                                      // ker2 + fac + f2
  { _Float16* w = wpre(P(77), 128, 128, 64, 64, 3);
    conv3(stream, s2[2], 64, H2, W2, w, P(78), s1[2], 128, 128, 1, 0, nullptr, 0); }// upconv_11
  pshuf(s1[2], s1[3], 32, H2, W2);                                                  // -> [HW1][32]
  resb(s1[3], s1[4], s1[5], 79, 32, H1, W1);                                        // upconv_12
  resb(s1[5], s1[4], s1[3], 83, 32, H1, W1);                                        // upconv_13

  // ---- decoder: scale 1 + final ----
  facL(s1[3], kf1, 61, f1, s1[2], 32, H1, W1);                                      // ker1 + fac + f1
  { _Float16* w = wpre(P(87), 4, 16, 32, 32, 3);
    conv3(stream, s1[2], 32, H1, W1, w, P(88), d_out, 4, 16, 1, 1, x, 1); }         // final+lrelu+x
}
